// SparseMixtureOfExperts_51032801411474
// MI455X (gfx1250) — compile-verified, matrix operand order checked by reading, not run on hardware
//
#include <hip/hip_runtime.h>
#include <hip/hip_bf16.h>

// ---------------------------------------------------------------------------
// Sparse MoE forward for MI455X (gfx1250, wave32, WMMA), top-2 routed.
//   router:  h = relu(x@rw1+rb1); probs = softmax(h@rw2+rb2); top-2 per token
//   out[n]  = sum_{e in top2(n)} probs[n,e] * ( relu(x[n]@ew1[e]+eb1[e]) @ ew2[e] + eb2[e] )
// Expert FFNs run ONLY on routed tokens (4x FLOP reduction vs dense):
//   - per-expert token lists built with atomics, padded to 128-row tiles
//   - layer-1 GEMM gathers token rows indirectly (index load outside K-loop)
//   - layer-2 GEMM scatters gate*(acc+bias) with global_atomic_add_f32
//     (2 commutative fp32 adds per token -> bit-deterministic output)
//   - fixed worst-case grids; blocks early-exit on device-side tile counts
// All GEMMs: v_wmma_f32_16x16x32_bf16, f32 accumulate, double-buffered LDS.
// ---------------------------------------------------------------------------

#define E_EXP 8
#define DIM   1024
#define HID   4096
#define NTOK  4096

typedef __attribute__((ext_vector_type(16))) __bf16 v16bf;
typedef __attribute__((ext_vector_type(8)))  __bf16 v8bf;
typedef __attribute__((ext_vector_type(8)))  float  v8f;
typedef __attribute__((ext_vector_type(4)))  float  v4f;

static __device__ __forceinline__ __bf16 f2bf(float f) {
  unsigned u = __builtin_bit_cast(unsigned, f);
  u += 0x7FFFu + ((u >> 16) & 1u);           // round-to-nearest-even
  unsigned short h = (unsigned short)(u >> 16);
  return __builtin_bit_cast(__bf16, h);
}
static __device__ __forceinline__ float bf2f(__bf16 b) {
  unsigned u = ((unsigned)__builtin_bit_cast(unsigned short, b)) << 16;
  return __builtin_bit_cast(float, u);
}

// ---------------------------------------------------------------------------
// Zero d_out (atomic-scatter target) and the expert counters.
// ---------------------------------------------------------------------------
__global__ __launch_bounds__(256) void zero_out_counts(
    float* __restrict__ out, int* __restrict__ count) {
  int i = (blockIdx.x * 256 + threadIdx.x) * 4;
  if (i + 4 <= NTOK * DIM) {
    v4f z = {0.0f, 0.0f, 0.0f, 0.0f};
    *reinterpret_cast<v4f*>(out + i) = z;
  }
  if (blockIdx.x == 0 && threadIdx.x < 2 * E_EXP) count[threadIdx.x] = 0;
}

// ---------------------------------------------------------------------------
// x: fp32 -> bf16
// ---------------------------------------------------------------------------
__global__ __launch_bounds__(256) void cvt_f32_to_bf16(
    const float* __restrict__ in, __bf16* __restrict__ out, int n) {
  int i = (blockIdx.x * 256 + threadIdx.x) * 8;
  if (i + 8 <= n) {
    v4f a = *reinterpret_cast<const v4f*>(in + i);
    v4f b = *reinterpret_cast<const v4f*>(in + i + 4);
    v8bf o;
    o[0] = f2bf(a[0]); o[1] = f2bf(a[1]); o[2] = f2bf(a[2]); o[3] = f2bf(a[3]);
    o[4] = f2bf(b[0]); o[5] = f2bf(b[1]); o[6] = f2bf(b[2]); o[7] = f2bf(b[3]);
    *reinterpret_cast<v8bf*>(out + i) = o;
  }
}

// ---------------------------------------------------------------------------
// Weights: fp32 [K,N] row-major -> bf16 [N,K] row-major (convert + transpose)
// grid = (K/32, N/32), block = 256
// ---------------------------------------------------------------------------
__global__ __launch_bounds__(256) void cvt_transpose_bf16(
    const float* __restrict__ src, __bf16* __restrict__ dst, int K, int N) {
  __shared__ __bf16 t[32][34];                 // 68B row stride (4B aligned)
  const int k0 = blockIdx.x * 32;
  const int n0 = blockIdx.y * 32;
  {
    const int r = threadIdx.x >> 5;            // 0..7
    const int c = threadIdx.x & 31;            // 0..31
#pragma unroll
    for (int i = 0; i < 4; ++i) {
      int kk = r + i * 8;
      t[c][kk] = f2bf(src[(size_t)(k0 + kk) * N + n0 + c]);
    }
  }
  __syncthreads();
  {
    const int rr = threadIdx.x >> 4;           // 0..15 (n within tile)
    const int cc = threadIdx.x & 15;           // 0..15 (k pair)
#pragma unroll
    for (int i = 0; i < 2; ++i) {
      int n = rr + i * 16;
      unsigned v = *reinterpret_cast<const unsigned*>(&t[n][cc * 2]);
      *reinterpret_cast<unsigned*>(&dst[(size_t)(n0 + n) * K + k0 + cc * 2]) = v;
    }
  }
}

// ---------------------------------------------------------------------------
// Tiled WMMA GEMM:  C = epilogue( A' @ Bt^T + bias ),  A'[m] = A[rowmap(m)]
// BM=128, BN=128, BK=64; 256 threads = 8 wave32; wave tile 32x64 (2x4 WMMA).
// MODE 0: dense A rows;      Cout(bf16)[m][n] = relu(acc + bias)
// MODE 1: A rows = tok[m];   Cout(bf16)[m][n] = relu(acc + bias)   (+early exit)
// MODE 2: dense A rows;      atomicAdd(&Cout(f32)[tok[m]][n], gate[m]*(acc+bias))
//                            (+early exit)
// mcount: device ptr to padded row count (MODE 1/2). M,N mult of 128, K of 64.
// ---------------------------------------------------------------------------
template <int MODE>
__global__ __launch_bounds__(256) void moe_gemm(
    const __bf16* __restrict__ A,
    const __bf16* __restrict__ Bt,
    const float*  __restrict__ bias,
    void* __restrict__ Cout,
    const int*   __restrict__ tok,
    const float* __restrict__ gatev,
    const int*   __restrict__ mcount,
    int Nn, int K) {
  constexpr int BM = 128, BN = 128, BK = 64, PAD = 8, LDK = BK + PAD;
  const int bm = blockIdx.x * BM;
  if (MODE != 0) {
    if (bm >= *mcount) return;                 // routed-token early exit
  }
  __shared__ __bf16 sA[2][BM][LDK];
  __shared__ __bf16 sB[2][BN][LDK];

  const int tid  = threadIdx.x;
  const int lane = tid & 31;
  const int wid  = tid >> 5;
  const int wm   = wid & 3;        // 4 x 32-row wave slabs
  const int wn   = wid >> 2;       // 2 x 64-col wave slabs
  const int bn   = blockIdx.y * BN;

  const int lmod  = lane & 15;
  const int khalf = (lane < 16) ? 0 : 8;
  const int mrow_off = (lane >= 16) ? 8 : 0;

  // staging: 128x64 bf16 = 1024 x v8bf chunks; 4 per thread per matrix
  const int s_r = tid >> 3;                    // 0..31 (+32*i)
  const int s_c = (tid & 7) * 8;               // 0..56

  // A row sources (indirect for MODE 1), resolved once outside the K loop
  int rowsrc[4];
#pragma unroll
  for (int i = 0; i < 4; ++i) {
    int r = bm + s_r + i * 32;
    rowsrc[i] = (MODE == 1) ? tok[r] : r;
  }

  v8bf ra[4], rb[4];
  auto gload = [&](int k0) {
#pragma unroll
    for (int i = 0; i < 4; ++i) {
      ra[i] = *reinterpret_cast<const v8bf*>(A  + (size_t)rowsrc[i] * K + k0 + s_c);
      rb[i] = *reinterpret_cast<const v8bf*>(Bt + (size_t)(bn + s_r + i * 32) * K + k0 + s_c);
    }
  };
  auto lstore = [&](int buf) {
#pragma unroll
    for (int i = 0; i < 4; ++i) {
      int r = s_r + i * 32;
      *reinterpret_cast<v8bf*>(&sA[buf][r][s_c]) = ra[i];
      *reinterpret_cast<v8bf*>(&sB[buf][r][s_c]) = rb[i];
    }
  };

  v8f acc[2][4] = {};

  gload(0);
  lstore(0);
  __syncthreads();

  const int nk = K / BK;
  for (int t = 0; t < nk; ++t) {
    const int buf = t & 1;
    if (t + 1 < nk) gload((t + 1) * BK);       // in flight during WMMA below
#pragma unroll
    for (int ks = 0; ks < 2; ++ks) {           // two K=32 WMMA steps per tile
      v16bf af[2], bfg[4];
#pragma unroll
      for (int mt = 0; mt < 2; ++mt) {
        const __bf16* p = &sA[buf][wm * 32 + mt * 16 + lmod][ks * 32];
        union { v16bf v; v8bf h[2]; } u;
        u.h[0] = *reinterpret_cast<const v8bf*>(p + khalf);
        u.h[1] = *reinterpret_cast<const v8bf*>(p + khalf + 16);
        af[mt] = u.v;
      }
#pragma unroll
      for (int nt = 0; nt < 4; ++nt) {
        const __bf16* p = &sB[buf][wn * 64 + nt * 16 + lmod][ks * 32];
        union { v16bf v; v8bf h[2]; } u;
        u.h[0] = *reinterpret_cast<const v8bf*>(p + khalf);
        u.h[1] = *reinterpret_cast<const v8bf*>(p + khalf + 16);
        bfg[nt] = u.v;
      }
#pragma unroll
      for (int mt = 0; mt < 2; ++mt)
#pragma unroll
        for (int nt = 0; nt < 4; ++nt)
          acc[mt][nt] = __builtin_amdgcn_wmma_f32_16x16x32_bf16(
              false, af[mt], false, bfg[nt], (short)0, acc[mt][nt], false, false);
    }
    if (t + 1 < nk) lstore(buf ^ 1);
    __syncthreads();
  }

  // epilogue: C/D layout: VGPR j -> row j (lanes 0-15) / j+8 (lanes 16-31), col = lane%16
#pragma unroll
  for (int mt = 0; mt < 2; ++mt) {
    int rowi[8];
    float gv[8];
    if (MODE == 2) {
#pragma unroll
      for (int j = 0; j < 8; ++j) {
        int m = bm + wm * 32 + mt * 16 + j + mrow_off;
        rowi[j] = tok[m];
        gv[j]   = gatev[m];
      }
    }
#pragma unroll
    for (int nt = 0; nt < 4; ++nt) {
      int n = bn + wn * 64 + nt * 16 + lmod;
      float bv = bias[n];
#pragma unroll
      for (int j = 0; j < 8; ++j) {
        float v = acc[mt][nt][j] + bv;
        if (MODE == 2) {
          atomicAdd((float*)Cout + (size_t)rowi[j] * Nn + n, gv[j] * v);
        } else {
          int m = bm + wm * 32 + mt * 16 + j + mrow_off;
          v = v > 0.0f ? v : 0.0f;
          ((__bf16*)Cout)[(size_t)m * Nn + n] = f2bf(v);
        }
      }
    }
  }
}

// ---------------------------------------------------------------------------
// Router tail: logits = hb @ rw2 + rb2 ; softmax ; top-2 ->
// per-expert token lists (tok[e][i], gate[e][i]) built with atomics.
// ---------------------------------------------------------------------------
__global__ __launch_bounds__(256) void router_topk_kernel(
    const __bf16* __restrict__ hb, const float* __restrict__ rw2,
    const float* __restrict__ rb2, int* __restrict__ count,
    int* __restrict__ tok, float* __restrict__ gateg) {
  int n = blockIdx.x * 256 + threadIdx.x;
  if (n >= NTOK) return;
  float acc[E_EXP];
#pragma unroll
  for (int e = 0; e < E_EXP; ++e) acc[e] = rb2[e];
  const __bf16* row = hb + (size_t)n * DIM;
  for (int d = 0; d < DIM; ++d) {
    float x = bf2f(row[d]);
    const float* w = rw2 + (size_t)d * E_EXP;
#pragma unroll
    for (int e = 0; e < E_EXP; ++e) acc[e] += x * w[e];
  }
  float mx = acc[0];
#pragma unroll
  for (int e = 1; e < E_EXP; ++e) mx = acc[e] > mx ? acc[e] : mx;
  float p[E_EXP], s = 0.0f;
#pragma unroll
  for (int e = 0; e < E_EXP; ++e) { p[e] = __expf(acc[e] - mx); s += p[e]; }
  float inv = 1.0f / s;
#pragma unroll
  for (int e = 0; e < E_EXP; ++e) p[e] *= inv;
  int i1 = 0;
#pragma unroll
  for (int e = 1; e < E_EXP; ++e) if (p[e] > p[i1]) i1 = e;
  int i2 = (i1 == 0) ? 1 : 0;
#pragma unroll
  for (int e = 0; e < E_EXP; ++e) if (e != i1 && p[e] > p[i2]) i2 = e;

  int q1 = atomicAdd(&count[i1], 1);
  tok[i1 * NTOK + q1]   = n;
  gateg[i1 * NTOK + q1] = p[i1];
  int q2 = atomicAdd(&count[i2], 1);
  tok[i2 * NTOK + q2]   = n;
  gateg[i2 * NTOK + q2] = p[i2];
}

// ---------------------------------------------------------------------------
// Pad each expert list up to a multiple of BM=128 with (token 0, gate 0),
// and publish padded counts at count[8+e]. Single block.
// ---------------------------------------------------------------------------
__global__ __launch_bounds__(256) void pad_lists_kernel(
    int* __restrict__ count, int* __restrict__ tok, float* __restrict__ gateg) {
#pragma unroll
  for (int e = 0; e < E_EXP; ++e) {
    int c  = count[e];
    int pc = (c + 127) & ~127;
    for (int i = c + threadIdx.x; i < pc; i += 256) {
      tok[e * NTOK + i]   = 0;
      gateg[e * NTOK + i] = 0.0f;
    }
    if (threadIdx.x == 0) count[E_EXP + e] = pc;
  }
}

// ---------------------------------------------------------------------------
extern "C" void kernel_launch(void* const* d_in, const int* in_sizes, int n_in,
                              void* d_out, int out_size, void* d_ws, size_t ws_size,
                              hipStream_t stream) {
  (void)in_sizes; (void)n_in; (void)out_size; (void)ws_size;
  const float* x   = (const float*)d_in[0];
  const float* rw1 = (const float*)d_in[1];
  const float* rb1 = (const float*)d_in[2];
  const float* rw2 = (const float*)d_in[3];
  const float* rb2 = (const float*)d_in[4];
  const float* ew1 = (const float*)d_in[5];
  const float* eb1 = (const float*)d_in[6];
  const float* ew2 = (const float*)d_in[7];
  const float* eb2 = (const float*)d_in[8];
  float* out = (float*)d_out;

  // Workspace carve-up (~67 MB total)
  char* ws = (char*)d_ws;
  __bf16* xb    = (__bf16*)(ws);                        //  8 MB: x bf16 [N,D]
  __bf16* hb    = (__bf16*)(ws + (size_t)(8u  << 20));  //  8 MB: router hidden
  __bf16* rw1t  = (__bf16*)(ws + (size_t)(16u << 20));  //  2 MB: rw1^T bf16 [D,D]
  __bf16* w1t   = (__bf16*)(ws + (size_t)(18u << 20));  //  8 MB: ew1[e]^T [H,D]
  __bf16* w2t   = (__bf16*)(ws + (size_t)(26u << 20));  //  8 MB: ew2[e]^T [D,H]
  __bf16* ehb   = (__bf16*)(ws + (size_t)(34u << 20));  // 32 MB: gathered hidden [N,H]
  int*    tok   = (int*)   (ws + (size_t)(66u << 20));  // 128 KB: token lists [E,N]
  float*  gateg = (float*) (ws + (size_t)(66u << 20) + 128 * 1024); // 128 KB gates
  int*    count = (int*)   (ws + (size_t)(66u << 20) + 256 * 1024); // [0..7] raw, [8..15] padded

  // 0) zero scatter target + counters
  zero_out_counts<<<(NTOK * DIM) / (256 * 4), 256, 0, stream>>>(out, count);

  // 1) x -> bf16
  cvt_f32_to_bf16<<<(NTOK * DIM) / (256 * 8), 256, 0, stream>>>(x, xb, NTOK * DIM);

  // 2) router layer 1: hb = relu(xb @ rw1 + rb1)
  cvt_transpose_bf16<<<dim3(DIM / 32, DIM / 32), 256, 0, stream>>>(rw1, rw1t, DIM, DIM);
  moe_gemm<0><<<dim3(NTOK / 128, DIM / 128), 256, 0, stream>>>(
      xb, rw1t, rb1, (void*)hb, nullptr, nullptr, nullptr, DIM, DIM);

  // 3) router tail: softmax + top-2 -> per-expert token lists
  router_topk_kernel<<<NTOK / 256, 256, 0, stream>>>(hb, rw2, rb2, count, tok, gateg);
  pad_lists_kernel<<<1, 256, 0, stream>>>(count, tok, gateg);

  // 4) experts: only routed tokens (worst-case grids, device-side early exit)
  for (int e = 0; e < E_EXP; ++e) {
    cvt_transpose_bf16<<<dim3(DIM / 32, HID / 32), 256, 0, stream>>>(
        ew1 + (size_t)e * DIM * HID, w1t, DIM, HID);
    cvt_transpose_bf16<<<dim3(HID / 32, DIM / 32), 256, 0, stream>>>(
        ew2 + (size_t)e * HID * DIM, w2t, HID, DIM);
    // layer 1 (gather): ehb[i] = relu(xb[tok[e][i]] @ ew1[e] + eb1[e])
    moe_gemm<1><<<dim3(NTOK / 128, HID / 128), 256, 0, stream>>>(
        xb, w1t, eb1 + (size_t)e * HID, (void*)ehb,
        tok + (size_t)e * NTOK, nullptr, count + E_EXP + e, HID, DIM);
    // layer 2 (scatter): out[tok[e][i]] += gate * (ehb[i] @ ew2[e] + eb2[e])
    moe_gemm<2><<<dim3(NTOK / 128, DIM / 128), 256, 0, stream>>>(
        ehb, w2t, eb2 + (size_t)e * DIM, (void*)out,
        tok + (size_t)e * NTOK, gateg + (size_t)e * NTOK, count + E_EXP + e, DIM, HID);
  }
}
